// NetAmazon_GAT_71768903516554
// MI455X (gfx1250) — compile-verified, matrix operand order checked by reading, not run on hardware
//
#include <hip/hip_runtime.h>
#include <hip/hip_bf16.h>
#include <math.h>

// ---------------------------------------------------------------------------
// CDNA5 (gfx1250) GAT: bf16 WMMA GEMMs (register-blocked over N) +
// L2-resident edge softmax/aggregation passes.
// ---------------------------------------------------------------------------

typedef __attribute__((ext_vector_type(16))) __bf16 v16bf;
typedef __attribute__((ext_vector_type(8)))  float  v8f;

#define NEG_SLOPE 0.2f

__device__ __forceinline__ unsigned short f2bf(float f) {
  unsigned int u = __float_as_uint(f);
  u += 0x7FFFu + ((u >> 16) & 1u);   // round-to-nearest-even
  return (unsigned short)(u >> 16);
}

__device__ __forceinline__ void atomicMaxF(float* addr, float val) {
  if (val >= 0.0f) atomicMax((int*)addr, __float_as_int(val));
  else             atomicMin((unsigned int*)addr, __float_as_uint(val));
}

// ---------------------------- conversion kernels ---------------------------

__global__ void k_f32_to_bf16(const float* __restrict__ src,
                              unsigned short* __restrict__ dst, int n) {
  int gid = blockIdx.x * blockDim.x + threadIdx.x;
  if (gid < n) dst[gid] = f2bf(src[gid]);
}

// W [Fin, Fout] row-major fp32 -> Wt [FoutPad, Fin] bf16 (transposed, zero padded)
__global__ void k_convW(const float* __restrict__ W, unsigned short* __restrict__ Wt,
                        int Fin, int Fout, int FoutPad) {
  int gid = blockIdx.x * blockDim.x + threadIdx.x;
  int tot = FoutPad * Fin;
  if (gid >= tot) return;
  int o = gid / Fin;
  int k = gid - o * Fin;
  Wt[gid] = (o < Fout) ? f2bf(W[(size_t)k * Fout + o]) : (unsigned short)0;
}

// ------------------------------- WMMA GEMM ---------------------------------
// xp[N,Fout] = Xb[N,Fin] (bf16) @ Wt[FoutPad,Fin]^T (bf16), fp32 accumulate.
// One wave -> one 16-row strip across the FULL output width (TILES 16x16
// tiles held in registers) so the A fragment is loaded once per k-step and
// reused by every column tile: loads/wmma ~ 2 + 2/TILES.

template <int TILES>
__global__ __launch_bounds__(256)
void k_gemm_wmma(const unsigned short* __restrict__ Xb,
                 const unsigned short* __restrict__ Wt,
                 float* __restrict__ xp,
                 int Nrows, int Fin, int Fout) {
  const int lane = threadIdx.x & 31;
  const int wave = threadIdx.x >> 5;
  const int mt   = blockIdx.x * 8 + wave;      // M tile index (wave-uniform)
  const int m0   = mt * 16;
  if (m0 >= Nrows) return;                     // whole-wave exit: EXEC stays all-1s
  const int half = lane >> 4;                  // 0: K 0..7/16..23, 1: K 8..15/24..31
  const int idx  = lane & 15;

  int arowi = m0 + idx;
  if (arowi >= Nrows) arowi = Nrows - 1;       // clamp (N%16==0 in practice)
  const unsigned short* arow = Xb + (size_t)arowi * Fin;
  const unsigned short* brow = Wt + (size_t)idx * Fin;
  const size_t brow_tile_stride = (size_t)16 * Fin;   // next column tile

  v8f acc[TILES];
  #pragma unroll
  for (int t = 0; t < TILES; ++t)
    acc[t] = (v8f){0.f, 0.f, 0.f, 0.f, 0.f, 0.f, 0.f, 0.f};

  for (int k0 = 0; k0 < Fin; k0 += 32) {
    union { v16bf v; uint4 q[2]; } A;
    A.q[0] = *reinterpret_cast<const uint4*>(arow + k0 + 8 * half);
    A.q[1] = *reinterpret_cast<const uint4*>(arow + k0 + 16 + 8 * half);
    #pragma unroll
    for (int t = 0; t < TILES; ++t) {
      union { v16bf v; uint4 q[2]; } B;
      const unsigned short* bp = brow + (size_t)t * brow_tile_stride + k0;
      B.q[0] = *reinterpret_cast<const uint4*>(bp + 8 * half);
      B.q[1] = *reinterpret_cast<const uint4*>(bp + 16 + 8 * half);
      acc[t] = __builtin_amdgcn_wmma_f32_16x16x32_bf16(
          /*neg_a=*/false, A.v, /*neg_b=*/false, B.v,
          /*c_mod=*/(short)0, acc[t], /*reuse_a=*/false, /*reuse_b=*/false);
    }
  }

  #pragma unroll
  for (int t = 0; t < TILES; ++t) {
    const int col = t * 16 + idx;
    if (col < Fout) {
      #pragma unroll
      for (int r = 0; r < 8; ++r) {
        int row = m0 + r + 8 * half;
        if (row < Nrows) xp[(size_t)row * Fout + col] = acc[t][r];
      }
    }
  }
}

// ----------------------------- attention logits ----------------------------

// alpha_src[n,h] = sum_c xp[n,h*C+c] * a_s[h*C+c]   (and same for dst)
__global__ void k_alpha(const float* __restrict__ xp,
                        const float* __restrict__ a_s, const float* __restrict__ a_d,
                        float* __restrict__ asrc, float* __restrict__ adst,
                        int N, int H, int C) {
  int gid = blockIdx.x * blockDim.x + threadIdx.x;
  if (gid >= N * H) return;
  int n = gid / H;
  int h = gid - n * H;
  const float* row = xp + (size_t)n * H * C + h * C;
  const float* vs = a_s + h * C;
  const float* vd = a_d + h * C;
  float ss = 0.f, sd = 0.f;
  for (int c = 0; c < C; ++c) { float v = row[c]; ss += v * vs[c]; sd += v * vd[c]; }
  asrc[gid] = ss;
  adst[gid] = sd;
}

__global__ void k_init(float* __restrict__ m, float* __restrict__ s,
                       float* __restrict__ out, int NH, int NF) {
  int gid = blockIdx.x * blockDim.x + threadIdx.x;
  if (gid < NF) out[gid] = 0.f;
  if (gid < NH) { m[gid] = -INFINITY; s[gid] = 0.f; }
}

// ------------------------------- edge passes -------------------------------

__device__ __forceinline__ void edge_sd(const long long* __restrict__ ei,
                                        int E, int e, int& s, int& d) {
  if (e < E) { s = (int)ei[e]; d = (int)ei[(size_t)E + e]; }
  else       { s = d = e - E; }     // self loops appended after real edges
}

__global__ void k_edge_max(const long long* __restrict__ ei, int E, int N,
                           const float* __restrict__ asrc, const float* __restrict__ adst,
                           float* __restrict__ m, int H) {
  int gid = blockIdx.x * blockDim.x + threadIdx.x;
  int tot = (E + N) * H;
  if (gid >= tot) return;
  int e = gid / H, h = gid - e * H;
  int s, d; edge_sd(ei, E, e, s, d);
  float v = asrc[(size_t)s * H + h] + adst[(size_t)d * H + h];
  v = (v > 0.f) ? v : NEG_SLOPE * v;
  atomicMaxF(&m[(size_t)d * H + h], v);
}

__global__ void k_edge_sum(const long long* __restrict__ ei, int E, int N,
                           const float* __restrict__ asrc, const float* __restrict__ adst,
                           const float* __restrict__ m, float* __restrict__ sbuf, int H) {
  int gid = blockIdx.x * blockDim.x + threadIdx.x;
  int tot = (E + N) * H;
  if (gid >= tot) return;
  int e = gid / H, h = gid - e * H;
  int s, d; edge_sd(ei, E, e, s, d);
  float v = asrc[(size_t)s * H + h] + adst[(size_t)d * H + h];
  v = (v > 0.f) ? v : NEG_SLOPE * v;
  float ez = expf(v - m[(size_t)d * H + h]);
  atomicAdd(&sbuf[(size_t)d * H + h], ez);
}

// edge-major flattening: consecutive lanes cover consecutive channels of one
// edge -> coalesced xp[src] gather and coalesced atomics on out[dst]
__global__ void k_edge_aggr(const long long* __restrict__ ei, int E, int N,
                            const float* __restrict__ asrc, const float* __restrict__ adst,
                            const float* __restrict__ m, const float* __restrict__ sbuf,
                            const float* __restrict__ xp, float* __restrict__ out,
                            int H, int C) {
  const int F = H * C;
  int gid = blockIdx.x * blockDim.x + threadIdx.x;
  int tot = (E + N) * F;
  if (gid >= tot) return;
  int e = gid / F, f = gid - e * F;
  int h = f / C;
  int s, d; edge_sd(ei, E, e, s, d);
  float v = asrc[(size_t)s * H + h] + adst[(size_t)d * H + h];
  v = (v > 0.f) ? v : NEG_SLOPE * v;
  float ez = expf(v - m[(size_t)d * H + h]);
  float alpha = ez / (sbuf[(size_t)d * H + h] + 1e-16f);
  atomicAdd(&out[(size_t)d * F + f], xp[(size_t)s * F + f] * alpha);
}

// ------------------------------- epilogues ---------------------------------

// h = elu(out + b); write bf16 directly (next layer's GEMM input)
__global__ void k_elu_to_bf16(const float* __restrict__ out, const float* __restrict__ b,
                              unsigned short* __restrict__ Xb, int N, int F) {
  int gid = blockIdx.x * blockDim.x + threadIdx.x;
  if (gid >= N * F) return;
  int f = gid % F;
  float v = out[gid] + b[f];
  v = (v > 0.f) ? v : (expf(v) - 1.f);
  Xb[gid] = f2bf(v);
}

__global__ void k_logsoftmax(const float* __restrict__ out, const float* __restrict__ b,
                             float* __restrict__ y, int N, int CLS) {
  int n = blockIdx.x * blockDim.x + threadIdx.x;
  if (n >= N) return;
  float t[16];
  float mx = -INFINITY;
  for (int j = 0; j < CLS; ++j) { t[j] = out[(size_t)n * CLS + j] + b[j]; mx = fmaxf(mx, t[j]); }
  float sum = 0.f;
  for (int j = 0; j < CLS; ++j) sum += expf(t[j] - mx);
  float lse = logf(sum);
  for (int j = 0; j < CLS; ++j) y[(size_t)n * CLS + j] = t[j] - mx - lse;
}

// ------------------------------- host driver -------------------------------

static inline int nblk(long long total, int bs) { return (int)((total + bs - 1) / bs); }

static void run_layer(const unsigned short* Xb_in, unsigned short* Xb_next,
                      const float* W, const float* a_s, const float* a_d, const float* b,
                      unsigned short* Wt, float* xp, float* outb,
                      float* asrc, float* adst, float* mbuf, float* sbuf,
                      const long long* ei, int N, int E,
                      int Fin, int Fout, int FoutPad, int H, int C,
                      float* final_out, hipStream_t stream) {
  // weight convert + transpose to bf16
  k_convW<<<nblk((long long)FoutPad * Fin, 256), 256, 0, stream>>>(W, Wt, Fin, Fout, FoutPad);
  // projection GEMM on the WMMA pipe (register-blocked over the full width)
  int mtiles = (N + 15) / 16;
  dim3 g((mtiles + 7) / 8, 1, 1);
  if (FoutPad == 128)
    k_gemm_wmma<8><<<g, 256, 0, stream>>>(Xb_in, Wt, xp, N, Fin, Fout);
  else if (FoutPad == 64)
    k_gemm_wmma<4><<<g, 256, 0, stream>>>(Xb_in, Wt, xp, N, Fin, Fout);
  else
    k_gemm_wmma<1><<<g, 256, 0, stream>>>(Xb_in, Wt, xp, N, Fin, Fout);
  // attention logits
  k_alpha<<<nblk((long long)N * H, 256), 256, 0, stream>>>(xp, a_s, a_d, asrc, adst, N, H, C);
  // init reductions
  k_init<<<nblk((long long)N * Fout, 256), 256, 0, stream>>>(mbuf, sbuf, outb, N * H, N * Fout);
  // segment softmax + aggregate
  long long EH = (long long)(E + N) * H;
  k_edge_max <<<nblk(EH, 256), 256, 0, stream>>>(ei, E, N, asrc, adst, mbuf, H);
  k_edge_sum <<<nblk(EH, 256), 256, 0, stream>>>(ei, E, N, asrc, adst, mbuf, sbuf, H);
  long long EF = (long long)(E + N) * Fout;
  k_edge_aggr<<<nblk(EF, 256), 256, 0, stream>>>(ei, E, N, asrc, adst, mbuf, sbuf, xp, outb, H, C);
  // epilogue
  if (final_out) {
    k_logsoftmax<<<nblk(N, 256), 256, 0, stream>>>(outb, b, final_out, N, Fout);
  } else {
    k_elu_to_bf16<<<nblk((long long)N * Fout, 256), 256, 0, stream>>>(outb, b, Xb_next, N, Fout);
  }
}

extern "C" void kernel_launch(void* const* d_in, const int* in_sizes, int n_in,
                              void* d_out, int out_size, void* d_ws, size_t ws_size,
                              hipStream_t stream) {
  const float*     x   = (const float*)d_in[0];
  const long long* ei  = (const long long*)d_in[1];   // int64 edge_index [2,E]
  const float* W1  = (const float*)d_in[2];
  const float* a1s = (const float*)d_in[3];
  const float* a1d = (const float*)d_in[4];
  const float* b1  = (const float*)d_in[5];
  const float* W2  = (const float*)d_in[6];
  const float* a2s = (const float*)d_in[7];
  const float* a2d = (const float*)d_in[8];
  const float* b2  = (const float*)d_in[9];
  const float* W3  = (const float*)d_in[10];
  const float* a3s = (const float*)d_in[11];
  const float* a3d = (const float*)d_in[12];
  const float* b3  = (const float*)d_in[13];
  float* y = (float*)d_out;

  const int F0 = 256;
  const int N  = in_sizes[0] / F0;
  const int E  = in_sizes[1] / 2;

  // workspace carve-out (256B aligned)
  size_t cur = 0;
  char* base = (char*)d_ws;
  auto take = [&](size_t bytes) -> char* {
    cur = (cur + 255) & ~(size_t)255;
    char* p = base + cur;
    cur += bytes;
    return p;
  };
  unsigned short* Xb   = (unsigned short*)take((size_t)N * 256 * sizeof(unsigned short));
  unsigned short* Wt   = (unsigned short*)take((size_t)128 * 256 * sizeof(unsigned short));
  float*          xp   = (float*)take((size_t)N * 128 * sizeof(float));
  float*          outb = (float*)take((size_t)N * 128 * sizeof(float));
  float*          asrc = (float*)take((size_t)N * 8 * sizeof(float));
  float*          adst = (float*)take((size_t)N * 8 * sizeof(float));
  float*          mbuf = (float*)take((size_t)N * 8 * sizeof(float));
  float*          sbuf = (float*)take((size_t)N * 8 * sizeof(float));
  (void)ws_size; (void)n_in; (void)out_size;

  // layer 0 input -> bf16
  k_f32_to_bf16<<<nblk((long long)N * F0, 256), 256, 0, stream>>>(x, Xb, N * F0);

  // L1: 256 -> 8 heads x 16  (Fout=128)
  run_layer(Xb, Xb, W1, a1s, a1d, b1, Wt, xp, outb, asrc, adst, mbuf, sbuf,
            ei, N, E, 256, 128, 128, 8, 16, nullptr, stream);
  // L2: 128 -> 8 heads x 8   (Fout=64)
  run_layer(Xb, Xb, W2, a2s, a2d, b2, Wt, xp, outb, asrc, adst, mbuf, sbuf,
            ei, N, E, 128, 64, 64, 8, 8, nullptr, stream);
  // L3: 64 -> 1 head x 10    (Fout=10, padded to 16) + log_softmax
  run_layer(Xb, nullptr, W3, a3s, a3d, b3, Wt, xp, outb, asrc, adst, mbuf, sbuf,
            ei, N, E, 64, 10, 16, 1, 10, y, stream);
}
